// DenseHypercube_53171695125388
// MI455X (gfx1250) — matrix operand order repaired and checked by reading, not compiled, not run on hardware
//
#include <hip/hip_runtime.h>
#include <hip/hip_bf16.h>
#include <math.h>

typedef __attribute__((ext_vector_type(16))) _Float16 v16h;
typedef __attribute__((ext_vector_type(8)))  float    v8f;

#define NDIV      25
#define MARG_LO   0.01f
#define MARG_SPAN 0.98f

#define NB1 1024
#define TPB 256
#define WPB (TPB / 32)   // waves per block (wave32)

// CDNA5 hardware tanh (TRANS op). Builtin when available, else inline asm
// with explicit V_NOPs for the TRANS-op result hazard.
__device__ __forceinline__ float fast_tanh(float v) {
#if __has_builtin(__builtin_amdgcn_tanhf)
    return __builtin_amdgcn_tanhf(v);
#elif __has_builtin(__builtin_amdgcn_tanh_f32)
    return __builtin_amdgcn_tanh_f32(v);
#else
    float r;
    asm volatile("v_tanh_f32 %0, %1\n\tv_nop\n\tv_nop" : "=v"(r) : "v"(v));
    return r;
#endif
}

// ---------------------------------------------------------------------------
// Kernel 1: fused MLP (layer1 via v_wmma_f32_16x16x32_f16, layers 2/3 VALU),
// writes e[N][2] and per-block min/max partials.
// ---------------------------------------------------------------------------
__global__ __launch_bounds__(TPB) void mlp_wmma_kernel(
    const float* __restrict__ x,
    const float* __restrict__ W1, const float* __restrict__ b1,
    const float* __restrict__ W2, const float* __restrict__ b2,
    const float* __restrict__ W3, const float* __restrict__ b3,
    float* __restrict__ e_out,        // [N][2]
    float* __restrict__ blk_part,     // [gridDim.x][4] = mn0,mx0,mn1,mx1
    int N)
{
    __shared__ float wsm[117];                 // W2(100) | b2(5) | W3(10) | b3(2)
    __shared__ float h1s[WPB][32][21];         // per-wave 32x20 tile (pad->21)
    __shared__ float red[TPB];

    const int tid  = threadIdx.x;
    const int lane = tid & 31;
    const int wave = tid >> 5;

    if (tid < 100)       wsm[tid] = W2[tid];
    else if (tid < 105)  wsm[tid] = b2[tid - 100];
    else if (tid < 115)  wsm[tid] = W3[tid - 105];
    else if (tid < 117)  wsm[tid] = b3[tid - 115];
    __syncthreads();

    const int col   = lane & 15;
    const int kbase = (lane < 16) ? 0 : 8;     // A-fragment K base (ISA layout)
    const int bk    = (lane < 16) ? 0 : 16;    // unused; the B fragment uses kbase too

    // Resident B fragments of W1^T padded 32x32 (cols 20..31 zero) + bias in C.
    v16h bf0, bf1;
    v8f  bias0, bias1;
    {
        const int c0 = col;
        const int c1 = 16 + col;
        #pragma unroll
        for (int h = 0; h < 16; ++h) {
            float w0 = (c0 < 20) ? W1[c0 * 32 + kbase + (h & 7) + ((h >> 3) << 4)] : 0.0f;
            float w1 = (c1 < 20) ? W1[c1 * 32 + kbase + (h & 7) + ((h >> 3) << 4)] : 0.0f;
            bf0[h] = (_Float16)w0;
            bf1[h] = (_Float16)w1;
        }
        float bb0 = (c0 < 20) ? b1[c0] : 0.0f;
        float bb1 = (c1 < 20) ? b1[c1] : 0.0f;
        #pragma unroll
        for (int v = 0; v < 8; ++v) { bias0[v] = bb0; bias1[v] = bb1; }
    }

    float mn0 = 1e30f, mx0 = -1e30f, mn1 = 1e30f, mx1 = -1e30f;

    const int nchunks    = (N + 31) >> 5;          // 32 rows per wave-chunk
    const int totalWaves = gridDim.x * WPB;

    for (int chunk = blockIdx.x * WPB + wave; chunk < nchunks; chunk += totalWaves)
    {
        const int base = chunk << 5;

        // ---- layer 1: two 16-row tiles, 2 WMMA each -----------------------
        #pragma unroll
        for (int t = 0; t < 2; ++t) {
            int row = base + t * 16 + col;
            int r   = (row < N) ? row : (N - 1);   // clamp keeps EXEC full
            const float* px = x + (size_t)r * 32;

            // A frag: halves 0..7 -> K kbase..+7, halves 8..15 -> K kbase+16..+23
            float4 f0 = *(const float4*)(px + kbase);
            float4 f1 = *(const float4*)(px + kbase + 4);
            float4 f2 = *(const float4*)(px + kbase + 16);
            float4 f3 = *(const float4*)(px + kbase + 20);
            v16h a;
            a[0]=(_Float16)f0.x;  a[1]=(_Float16)f0.y;  a[2]=(_Float16)f0.z;  a[3]=(_Float16)f0.w;
            a[4]=(_Float16)f1.x;  a[5]=(_Float16)f1.y;  a[6]=(_Float16)f1.z;  a[7]=(_Float16)f1.w;
            a[8]=(_Float16)f2.x;  a[9]=(_Float16)f2.y;  a[10]=(_Float16)f2.z; a[11]=(_Float16)f2.w;
            a[12]=(_Float16)f3.x; a[13]=(_Float16)f3.y; a[14]=(_Float16)f3.z; a[15]=(_Float16)f3.w;

            v8f c0 = __builtin_amdgcn_wmma_f32_16x16x32_f16(
                        false, a, false, bf0, (short)0, bias0, false, false);
            v8f c1 = __builtin_amdgcn_wmma_f32_16x16x32_f16(
                        false, a, false, bf1, (short)0, bias1, false, false);

            // C layout: lane<16 -> N=lane, M=v ; lane>=16 -> N=lane-16, M=8+v
            const int mbase = t * 16 + ((lane < 16) ? 0 : 8);
            #pragma unroll
            for (int v = 0; v < 8; ++v)
                h1s[wave][mbase + v][col] = fast_tanh(c0[v]);
            if (col < 4) {
                #pragma unroll
                for (int v = 0; v < 8; ++v)
                    h1s[wave][mbase + v][16 + col] = fast_tanh(c1[v]);
            }
        }
        __builtin_amdgcn_wave_barrier();   // keep LDS store->load program order

        // ---- layers 2 & 3: one point per lane from LDS --------------------
        {
            const int row = base + lane;
            float hv[20];
            #pragma unroll
            for (int k = 0; k < 20; ++k) hv[k] = h1s[wave][lane][k];

            float h2[5];
            #pragma unroll
            for (int j = 0; j < 5; ++j) {
                float s = wsm[100 + j];
                #pragma unroll
                for (int k = 0; k < 20; ++k)
                    s = fmaf(hv[k], wsm[j * 20 + k], s);
                h2[j] = fast_tanh(s);
            }
            float e0 = wsm[115], e1 = wsm[116];
            #pragma unroll
            for (int j = 0; j < 5; ++j) {
                e0 = fmaf(h2[j], wsm[105 + j], e0);
                e1 = fmaf(h2[j], wsm[110 + j], e1);
            }
            e0 = fast_tanh(e0);
            e1 = fast_tanh(e1);
            mn0 = fminf(mn0, e0); mx0 = fmaxf(mx0, e0);
            mn1 = fminf(mn1, e1); mx1 = fmaxf(mx1, e1);
            if (row < N)
                *(float2*)(e_out + (size_t)row * 2) = make_float2(e0, e1);
        }
        __builtin_amdgcn_wave_barrier();
    }

    // ---- block min/max reduction --------------------------------------
    float vals[4] = { mn0, mx0, mn1, mx1 };
    #pragma unroll
    for (int q = 0; q < 4; ++q) {
        const bool isMin = (q == 0 || q == 2);
        red[tid] = vals[q];
        __syncthreads();
        for (int s = TPB / 2; s > 0; s >>= 1) {
            if (tid < s)
                red[tid] = isMin ? fminf(red[tid], red[tid + s])
                                 : fmaxf(red[tid], red[tid + s]);
            __syncthreads();
        }
        if (tid == 0) blk_part[blockIdx.x * 4 + q] = red[0];
        __syncthreads();
    }
}

// ---------------------------------------------------------------------------
// Kernel 2: reduce block partials -> stats = {mn0, 0.98/(mx0-mn0), mn1, ...}
// ---------------------------------------------------------------------------
__global__ __launch_bounds__(256) void reduce_stats_kernel(
    const float* __restrict__ blk_part, int nblk, float* __restrict__ stats)
{
    __shared__ float s0[256], s1[256], s2[256], s3[256];
    const int tid = threadIdx.x;
    float mn0 = 1e30f, mx0 = -1e30f, mn1 = 1e30f, mx1 = -1e30f;
    for (int i = tid; i < nblk; i += 256) {
        mn0 = fminf(mn0, blk_part[i * 4 + 0]);
        mx0 = fmaxf(mx0, blk_part[i * 4 + 1]);
        mn1 = fminf(mn1, blk_part[i * 4 + 2]);
        mx1 = fmaxf(mx1, blk_part[i * 4 + 3]);
    }
    s0[tid] = mn0; s1[tid] = mx0; s2[tid] = mn1; s3[tid] = mx1;
    __syncthreads();
    for (int s = 128; s > 0; s >>= 1) {
        if (tid < s) {
            s0[tid] = fminf(s0[tid], s0[tid + s]);
            s1[tid] = fmaxf(s1[tid], s1[tid + s]);
            s2[tid] = fminf(s2[tid], s2[tid + s]);
            s3[tid] = fmaxf(s3[tid], s3[tid + s]);
        }
        __syncthreads();
    }
    if (tid == 0) {
        stats[0] = s0[0];
        stats[1] = MARG_SPAN / (s1[0] - s0[0]);
        stats[2] = s2[0];
        stats[3] = MARG_SPAN / (s3[0] - s2[0]);
    }
}

// ---------------------------------------------------------------------------
// Kernel 3: normalize, bin, gather b_m  (pure bandwidth)
// ---------------------------------------------------------------------------
__global__ __launch_bounds__(256) void bin_gather_kernel(
    const float* __restrict__ e, const float* __restrict__ stats,
    const float* __restrict__ b_m, float* __restrict__ y, int N)
{
    const int i = blockIdx.x * blockDim.x + threadIdx.x;
    if (i >= N) return;
    const float mn0 = stats[0], sc0 = stats[1];
    const float mn1 = stats[2], sc1 = stats[3];
    float2 ev = *(const float2*)(e + (size_t)i * 2);
    float u0 = (ev.x - mn0) * sc0 + MARG_LO;
    float u1 = (ev.y - mn1) * sc1 + MARG_LO;
    int d0 = (int)floorf(u0 * (float)NDIV);
    int d1 = (int)floorf(u1 * (float)NDIV);
    d0 = min(max(d0, 0), NDIV - 1);
    d1 = min(max(d1, 0), NDIV - 1);
    y[i] = b_m[d0 * NDIV + d1];
}

// ---------------------------------------------------------------------------
extern "C" void kernel_launch(void* const* d_in, const int* in_sizes, int n_in,
                              void* d_out, int out_size, void* d_ws, size_t ws_size,
                              hipStream_t stream)
{
    const float* x  = (const float*)d_in[0];
    const float* W1 = (const float*)d_in[1];
    const float* b1 = (const float*)d_in[2];
    const float* W2 = (const float*)d_in[3];
    const float* b2 = (const float*)d_in[4];
    const float* W3 = (const float*)d_in[5];
    const float* b3 = (const float*)d_in[6];
    const float* bm = (const float*)d_in[7];
    float* out = (float*)d_out;

    const int N = in_sizes[0] / 32;

    // workspace layout: e[N*2] | block partials[NB1*4] | stats[4]
    float* e_buf    = (float*)d_ws;
    float* blk_part = e_buf + (size_t)N * 2;
    float* stats    = blk_part + (size_t)NB1 * 4;

    mlp_wmma_kernel<<<NB1, TPB, 0, stream>>>(x, W1, b1, W2, b2, W3, b3,
                                             e_buf, blk_part, N);
    reduce_stats_kernel<<<1, 256, 0, stream>>>(blk_part, NB1, stats);
    const int nb3 = (N + 255) / 256;
    bin_gather_kernel<<<nb3, 256, 0, stream>>>(e_buf, stats, bm, out, N);
}
